// DisBlock_8564164788958
// MI455X (gfx1250) — matrix operand order repaired and checked by reading, not compiled, not run on hardware
//
#include <hip/hip_runtime.h>
#include <hip/hip_bf16.h>

// ---------------------------------------------------------------------------
// DisBlock (Swin-style transformer block) for gfx1250 / MI455X.
// bf16 WMMA (v_wmma_f32_16x16x32_bf16) for all GEMMs, fp32 VALU for LN /
// softmax / GELU. wave32 throughout.
// ---------------------------------------------------------------------------

#define BB   128
#define NTOK 256
#define CH   512
#define NH   8
#define DH   64
#define HID  2048
#define MROW (BB * NTOK)     // 32768
#define SCALE 0.125f         // 64^-0.5

typedef __attribute__((ext_vector_type(16))) __bf16 v16bf;
typedef __attribute__((ext_vector_type(8)))  float  v8f;
typedef unsigned int u32;

union Frag { v16bf v; u32 u[8]; __bf16 h[16]; };

__device__ __forceinline__ __bf16 f2bf(float f) {
    union { float f; u32 u; } a; a.f = f;
    u32 r = a.u + 0x7FFFu + ((a.u >> 16) & 1u);   // round-to-nearest-even
    unsigned short hs = (unsigned short)(r >> 16);
    return __builtin_bit_cast(__bf16, hs);
}

__device__ __forceinline__ __bf16 bf_lo(u32 w) {
    return __builtin_bit_cast(__bf16, (unsigned short)(w & 0xFFFFu));
}
__device__ __forceinline__ __bf16 bf_hi(u32 w) {
    return __builtin_bit_cast(__bf16, (unsigned short)(w >> 16));
}

__device__ __forceinline__ v8f wmma_bf16(v16bf a, v16bf b, v8f c) {
    return __builtin_amdgcn_wmma_f32_16x16x32_bf16(false, a, false, b,
                                                   (short)0, c, false, false);
}

// ---------------------------------------------------------------------------
// fp32 -> bf16 convert (weights)
// ---------------------------------------------------------------------------
__global__ __launch_bounds__(256) void cvt_kernel(const float* __restrict__ in,
                                                  __bf16* __restrict__ out, int n) {
    int i = blockIdx.x * 256 + threadIdx.x;
    if (i < n) out[i] = f2bf(in[i]);
}

// ---------------------------------------------------------------------------
// bias[h][n][m] = rp_table[rel_index[n*256+m]][h]
// ---------------------------------------------------------------------------
__global__ __launch_bounds__(256) void bias_kernel(const float* __restrict__ rp,
                                                   const int* __restrict__ ridx,
                                                   float* __restrict__ biasH) {
    int i = blockIdx.x * 256 + threadIdx.x;     // over NTOK*NTOK
    if (i < NTOK * NTOK) {
        int t = ridx[i];
        #pragma unroll
        for (int h = 0; h < NH; ++h)
            biasH[(size_t)h * NTOK * NTOK + i] = rp[t * NH + h];
    }
}

// ---------------------------------------------------------------------------
// LayerNorm (+ optional noise injection) -> bf16.  One 128-thread block/row.
// ---------------------------------------------------------------------------
__global__ __launch_bounds__(128) void ln_kernel(const float* __restrict__ x,
                                                 const float* __restrict__ g,
                                                 const float* __restrict__ bt,
                                                 const float* __restrict__ noise,
                                                 const float* __restrict__ nsp,
                                                 __bf16* __restrict__ out) {
    int row = blockIdx.x;
    const float* xr = x + (size_t)row * CH;
    int tid = threadIdx.x, lane = tid & 31, wv = tid >> 5;
    float v[4];
    #pragma unroll
    for (int i = 0; i < 4; ++i) v[i] = xr[tid + i * 128];
    float s = v[0] + v[1] + v[2] + v[3];
    #pragma unroll
    for (int off = 16; off; off >>= 1) s += __shfl_xor(s, off);
    __shared__ float red[4];
    if (!lane) red[wv] = s;
    __syncthreads();
    float mean = (red[0] + red[1] + red[2] + red[3]) * (1.f / CH);
    float s2 = 0.f;
    #pragma unroll
    for (int i = 0; i < 4; ++i) { float d = v[i] - mean; s2 += d * d; }
    #pragma unroll
    for (int off = 16; off; off >>= 1) s2 += __shfl_xor(s2, off);
    __syncthreads();
    if (!lane) red[wv] = s2;
    __syncthreads();
    float rstd = rsqrtf((red[0] + red[1] + red[2] + red[3]) * (1.f / CH) + 1e-5f);
    float nz = noise ? noise[row] * nsp[0] : 0.f;
    #pragma unroll
    for (int i = 0; i < 4; ++i) {
        int c = tid + i * 128;
        out[(size_t)row * CH + c] = f2bf((v[i] - mean) * rstd * g[c] + bt[c] + nz);
    }
}

// ---------------------------------------------------------------------------
// Generic bf16 WMMA GEMM:  out[M,Nc] = A[M,K] x W[Nc,K]^T  (+ epilogue)
//   MODE 0: scatter to q/k/v [B,H,N,D] bf16 (q scaled)
//   MODE 1: fp32 out = acc + bias[c] + resid[r,c]
//   MODE 2: bf16 out = gelu(acc + bias[c])
// Block: 256 thr = 8 waves; wave tile 32x64; macro tile 128x128; K-step 32.
// ---------------------------------------------------------------------------
template <int MODE>
__global__ __launch_bounds__(256, 1) void gemm_bf16(
    const __bf16* __restrict__ A, const __bf16* __restrict__ Wt,
    int Ncols, int Kdim,
    const float* __restrict__ biasv, const float* __restrict__ resid,
    float* __restrict__ outf, __bf16* __restrict__ outb,
    __bf16* __restrict__ qo, __bf16* __restrict__ ko, __bf16* __restrict__ vo) {
    int lane = threadIdx.x & 31, wv = threadIdx.x >> 5;
    int wm = wv & 3, wn = wv >> 2;
    int m0 = blockIdx.x * 128 + wm * 32;
    int n0 = blockIdx.y * 128 + wn * 64;
    int l16 = lane & 15;
    int kb0 = (lane < 16) ? 0 : 8;
    int kb1 = (lane < 16) ? 16 : 24;
    int kbB = (lane < 16) ? 0 : 16;

    const u32* ar0 = (const u32*)(A + (size_t)(m0 + l16) * Kdim);
    const u32* ar1 = (const u32*)(A + (size_t)(m0 + 16 + l16) * Kdim);
    const u32* wr[4];
    #pragma unroll
    for (int t = 0; t < 4; ++t)
        wr[t] = (const u32*)(Wt + (size_t)(n0 + t * 16 + l16) * Kdim);

    v8f acc[2][4] = {};
    for (int k0 = 0; k0 < Kdim; k0 += 32) {
        Frag a0, a1, bw[4];
        #pragma unroll
        for (int i = 0; i < 4; ++i) {
            a0.u[i]     = ar0[(k0 + kb0 + 2 * i) >> 1];
            a0.u[i + 4] = ar0[(k0 + kb1 + 2 * i) >> 1];
            a1.u[i]     = ar1[(k0 + kb0 + 2 * i) >> 1];
            a1.u[i + 4] = ar1[(k0 + kb1 + 2 * i) >> 1];
        }
        #pragma unroll
        for (int t = 0; t < 4; ++t)
            #pragma unroll
            for (int i = 0; i < 8; ++i)
                bw[t].u[i] = wr[t][(k0 + (i < 4 ? kb0 + 2 * i : kb1 + 2 * (i - 4))) >> 1];
        __builtin_prefetch(ar0 + ((k0 + 64) >> 1), 0, 1);   // global_prefetch_b8
        __builtin_prefetch(ar1 + ((k0 + 64) >> 1), 0, 1);
        #pragma unroll
        for (int t = 0; t < 4; ++t) {
            acc[0][t] = wmma_bf16(a0.v, bw[t].v, acc[0][t]);
            acc[1][t] = wmma_bf16(a1.v, bw[t].v, acc[1][t]);
        }
    }

    #pragma unroll
    for (int i = 0; i < 2; ++i)
        #pragma unroll
        for (int t = 0; t < 4; ++t)
            #pragma unroll
            for (int j = 0; j < 8; ++j) {
                int r = m0 + i * 16 + j + ((lane >= 16) ? 8 : 0);
                int c = n0 + t * 16 + l16;
                float v = acc[i][t][j];
                if (MODE == 0) {
                    int s = c >> 9, rem = c & 511, hh = rem >> 6, d = rem & 63;
                    int bb = r >> 8, n = r & 255;
                    size_t idx = (((size_t)bb * NH + hh) * NTOK + n) * DH + d;
                    if (s == 0)      qo[idx] = f2bf(v * SCALE);
                    else if (s == 1) ko[idx] = f2bf(v);
                    else             vo[idx] = f2bf(v);
                } else if (MODE == 1) {
                    size_t idx = (size_t)r * Ncols + c;
                    outf[idx] = v + biasv[c] + resid[idx];
                } else {
                    float t2 = v + biasv[c];
                    outb[(size_t)r * Ncols + c] =
                        f2bf(0.5f * t2 * (1.f + erff(t2 * 0.70710678118f)));
                }
            }
}

// ---------------------------------------------------------------------------
// Attention: softmax(Q K^T + bias) V.  One block per (b, h, 128-row half);
// 8 waves, each owns a 16-row band.
// K and V^T are cooperatively staged in LDS once per block (coalesced b64
// global loads), then all WMMA B-fragments come from aligned dword ds reads.
//   Kl : [256][72]  bf16  (pad 64->72 : dword stride 36 -> low-conflict)
//   Vt : [64][264]  bf16  (transposed; dword stride 132 -> conflict-free)
//   P  : per-wave [16][264] bf16 softmax matrix
// ---------------------------------------------------------------------------
#define KSTR 72
#define VSTR 264
#define PSTR 264
#define KL_ELEMS (NTOK * KSTR)          // 18432
#define VT_ELEMS (DH * VSTR)            // 16896
#define ATTN_LDS_BYTES ((KL_ELEMS + VT_ELEMS + 8 * 16 * PSTR) * 2)

__global__ __launch_bounds__(256, 1) void attn_kernel(
    const __bf16* __restrict__ Q, const __bf16* __restrict__ Kt,
    const __bf16* __restrict__ V, const float* __restrict__ biasH,
    __bf16* __restrict__ Obnc) {
    extern __shared__ __bf16 smem[];
    __bf16* Kl = smem;
    __bf16* Vt = smem + KL_ELEMS;
    int lane = threadIdx.x & 31, wv = threadIdx.x >> 5;
    int bid = blockIdx.x;
    int half = bid & 1, bh = bid >> 1;
    int b = bh >> 3, h = bh & 7;
    int rbase = half * 128 + wv * 16;
    int l16 = lane & 15;
    int kb0 = (lane < 16) ? 0 : 8;
    int kb1 = (lane < 16) ? 16 : 24;
    int kbB = (lane < 16) ? 0 : 16;

    const __bf16* qhd = Q + (size_t)bh * NTOK * DH;
    const __bf16* khd = Kt + (size_t)bh * NTOK * DH;
    const __bf16* vhd = V + (size_t)bh * NTOK * DH;
    const float* bias = biasH + (size_t)h * NTOK * NTOK;

    // ---- cooperative staging: K (copy) and V (transpose) into LDS --------
    {
        const u32* kg = (const u32*)khd;
        const u32* vg = (const u32*)vhd;
        u32* kl = (u32*)Kl;                       // row stride 36 dwords
        int tid = threadIdx.x;
        #pragma unroll
        for (int it = 0; it < 16; ++it) {
            int id = it * 256 + tid;              // chunk = 4 bf16 along d
            int row = id >> 4, dc = id & 15;
            u32 ka = kg[row * 32 + dc * 2];
            u32 kb = kg[row * 32 + dc * 2 + 1];
            kl[row * (KSTR / 2) + dc * 2]     = ka;
            kl[row * (KSTR / 2) + dc * 2 + 1] = kb;
            u32 va = vg[row * 32 + dc * 2];
            u32 vb2 = vg[row * 32 + dc * 2 + 1];
            int d0 = dc * 4;
            Vt[(d0 + 0) * VSTR + row] = bf_lo(va);
            Vt[(d0 + 1) * VSTR + row] = bf_hi(va);
            Vt[(d0 + 2) * VSTR + row] = bf_lo(vb2);
            Vt[(d0 + 3) * VSTR + row] = bf_hi(vb2);
        }
    }
    __syncthreads();

    // ---- Q A-fragments (rows rbase..rbase+15; SCALE pre-folded into Q) ----
    Frag aq[2];
    {
        const u32* qrow = (const u32*)(qhd + (size_t)(rbase + l16) * DH);
        #pragma unroll
        for (int ks = 0; ks < 2; ++ks)
            #pragma unroll
            for (int i = 0; i < 4; ++i) {
                aq[ks].u[i]     = qrow[(ks * 32 + kb0 + 2 * i) >> 1];
                aq[ks].u[i + 4] = qrow[(ks * 32 + kb1 + 2 * i) >> 1];
            }
    }

    // ---- S = Q K^T  (16 x 256 band, f32 accum; B-frags from LDS) ----------
    v8f acc[16];
    #pragma unroll
    for (int ct = 0; ct < 16; ++ct) {
        v8f c = {};
        const u32* krow = (const u32*)Kl + (size_t)(ct * 16 + l16) * (KSTR / 2);
        #pragma unroll
        for (int ks = 0; ks < 2; ++ks) {
            Frag bk;
            #pragma unroll
            for (int i = 0; i < 8; ++i)
                bk.u[i] = krow[(ks * 32 + (i < 4 ? kb0 + 2 * i : kb1 + 2 * (i - 4))) >> 1];
            c = wmma_bf16(aq[ks].v, bk.v, c);
        }
        acc[ct] = c;
    }

    // ---- bias + row softmax (rows stripe across the 16-lane halves) -------
    float rinv[8];
    #pragma unroll
    for (int j = 0; j < 8; ++j) {
        int n = rbase + j + ((lane >= 16) ? 8 : 0);
        float mx = -3.4e38f;
        #pragma unroll
        for (int ct = 0; ct < 16; ++ct) {
            float t = acc[ct][j] + bias[(size_t)n * NTOK + ct * 16 + l16];
            acc[ct][j] = t;
            mx = fmaxf(mx, t);
        }
        #pragma unroll
        for (int off = 1; off < 16; off <<= 1) mx = fmaxf(mx, __shfl_xor(mx, off));
        float s = 0.f;
        #pragma unroll
        for (int ct = 0; ct < 16; ++ct) {
            float e = __expf(acc[ct][j] - mx);
            acc[ct][j] = e;
            s += e;
        }
        #pragma unroll
        for (int off = 1; off < 16; off <<= 1) s += __shfl_xor(s, off);
        rinv[j] = 1.f / s;
    }

    // ---- stage normalized P (bf16) in per-wave LDS region -----------------
    __bf16* P = smem + KL_ELEMS + VT_ELEMS + (size_t)wv * 16 * PSTR;
    #pragma unroll
    for (int j = 0; j < 8; ++j) {
        int r = j + ((lane >= 16) ? 8 : 0);
        #pragma unroll
        for (int ct = 0; ct < 16; ++ct)
            P[r * PSTR + ct * 16 + l16] = f2bf(acc[ct][j] * rinv[j]);
    }

    // ---- O = P x V  (16 x 64; A-frags from P, B-frags from Vt) ------------
    #pragma unroll
    for (int dt = 0; dt < 4; ++dt) {
        v8f c = {};
        const u32* prow = (const u32*)(P + (size_t)l16 * PSTR);
        const u32* vcol = (const u32*)(Vt + (size_t)(dt * 16 + l16) * VSTR);
        for (int mt = 0; mt < 8; ++mt) {
            Frag ap, bv;
            #pragma unroll
            for (int i = 0; i < 4; ++i) {
                ap.u[i]     = prow[(mt * 32 + kb0 + 2 * i) >> 1];
                ap.u[i + 4] = prow[(mt * 32 + kb1 + 2 * i) >> 1];
            }
            #pragma unroll
            for (int i = 0; i < 8; ++i)
                bv.u[i] = vcol[(mt * 32 + (i < 4 ? kb0 + 2 * i : kb1 + 2 * (i - 4))) >> 1];
            c = wmma_bf16(ap.v, bv.v, c);
        }
        #pragma unroll
        for (int j = 0; j < 8; ++j) {
            int n = rbase + j + ((lane >= 16) ? 8 : 0);
            Obnc[(size_t)(b * NTOK + n) * CH + h * DH + dt * 16 + l16] = f2bf(c[j]);
        }
    }
}

// ---------------------------------------------------------------------------
// Host launcher
// ---------------------------------------------------------------------------
extern "C" void kernel_launch(void* const* d_in, const int* in_sizes, int n_in,
                              void* d_out, int out_size, void* d_ws, size_t ws_size,
                              hipStream_t stream) {
    const float* x      = (const float*)d_in[0];
    const float* noise  = (const float*)d_in[1];
    const float* ns     = (const float*)d_in[2];
    const float* g1     = (const float*)d_in[3];
    const float* b1     = (const float*)d_in[4];
    const float* w_qkv  = (const float*)d_in[5];
    const float* w_proj = (const float*)d_in[6];
    const float* b_proj = (const float*)d_in[7];
    const float* rp     = (const float*)d_in[8];
    const float* g2     = (const float*)d_in[9];
    const float* b2     = (const float*)d_in[10];
    const float* w1     = (const float*)d_in[11];
    const float* b1m    = (const float*)d_in[12];
    const float* w2     = (const float*)d_in[13];
    const float* b2m    = (const float*)d_in[14];
    const int*   ridx   = (const int*)d_in[15];
    float* out = (float*)d_out;

    char* p = (char*)d_ws;
    auto carve = [&](size_t bytes) -> void* {
        void* r = (void*)p;
        p += (bytes + 255) & ~(size_t)255;
        return r;
    };
    __bf16* wqB   = (__bf16*)carve((size_t)3 * CH * CH * 2);
    __bf16* wpB   = (__bf16*)carve((size_t)CH * CH * 2);
    __bf16* w1B   = (__bf16*)carve((size_t)HID * CH * 2);
    __bf16* w2B   = (__bf16*)carve((size_t)CH * HID * 2);
    float*  biasH = (float*)carve((size_t)NH * NTOK * NTOK * 4);
    __bf16* h1    = (__bf16*)carve((size_t)MROW * CH * 2);
    __bf16* qb    = (__bf16*)carve((size_t)MROW * CH * 2);
    __bf16* kb    = (__bf16*)carve((size_t)MROW * CH * 2);
    __bf16* vb    = (__bf16*)carve((size_t)MROW * CH * 2);
    __bf16* obnc  = (__bf16*)carve((size_t)MROW * CH * 2);
    float*  xattn = (float*)carve((size_t)MROW * CH * 4);
    __bf16* h2    = (__bf16*)carve((size_t)MROW * CH * 2);
    __bf16* act   = (__bf16*)carve((size_t)MROW * HID * 2);

    // weight converts + bias table
    cvt_kernel<<<(3 * CH * CH + 255) / 256, 256, 0, stream>>>(w_qkv, wqB, 3 * CH * CH);
    cvt_kernel<<<(CH * CH + 255) / 256, 256, 0, stream>>>(w_proj, wpB, CH * CH);
    cvt_kernel<<<(HID * CH + 255) / 256, 256, 0, stream>>>(w1, w1B, HID * CH);
    cvt_kernel<<<(CH * HID + 255) / 256, 256, 0, stream>>>(w2, w2B, CH * HID);
    bias_kernel<<<(NTOK * NTOK + 255) / 256, 256, 0, stream>>>(rp, ridx, biasH);

    // LN1 + noise
    ln_kernel<<<MROW, 128, 0, stream>>>(x, g1, b1, noise, ns, h1);

    // QKV projection (M=32768, Nc=1536, K=512)
    gemm_bf16<0><<<dim3(MROW / 128, (3 * CH) / 128), 256, 0, stream>>>(
        h1, wqB, 3 * CH, CH, nullptr, nullptr, nullptr, nullptr, qb, kb, vb);

    // attention
    attn_kernel<<<BB * NH * 2, 256, ATTN_LDS_BYTES, stream>>>(qb, kb, vb, biasH, obnc);

    // output projection + residual -> xattn (fp32)
    gemm_bf16<1><<<dim3(MROW / 128, CH / 128), 256, 0, stream>>>(
        obnc, wpB, CH, CH, b_proj, x, xattn, nullptr, nullptr, nullptr, nullptr);

    // LN2
    ln_kernel<<<MROW, 128, 0, stream>>>(xattn, g2, b2, nullptr, nullptr, h2);

    // MLP up + GELU (Nc=2048, K=512)
    gemm_bf16<2><<<dim3(MROW / 128, HID / 128), 256, 0, stream>>>(
        h2, w1B, HID, CH, b1m, nullptr, nullptr, act, nullptr, nullptr, nullptr);

    // MLP down + bias + residual -> d_out (Nc=512, K=2048)
    gemm_bf16<1><<<dim3(MROW / 128, CH / 128), 256, 0, stream>>>(
        act, w2B, CH, HID, b2m, xattn, out, nullptr, nullptr, nullptr, nullptr);

    (void)in_sizes; (void)n_in; (void)out_size; (void)ws_size;
}
